// AttentionBlock_36000415875763
// MI455X (gfx1250) — matrix operand order repaired and checked by reading, not compiled, not run on hardware
//
#include <hip/hip_runtime.h>

// ---------------------------------------------------------------------------
// SAGAN attention block for MI455X (gfx1250, wave32, WMMA bf16 16x16x32)
// B=4, C=256, H=W=64 -> N=4096, Co=32
// ---------------------------------------------------------------------------

typedef __attribute__((ext_vector_type(16))) __bf16 v16bf;
typedef __attribute__((ext_vector_type(8)))  __bf16 v8bf;
typedef __attribute__((ext_vector_type(8)))  float  v8f;

#define NB    4
#define CIN   256
#define CQK   32
#define NTOK  4096

__device__ __forceinline__ v8f wmma_bf16(v16bf a, v16bf b, v8f c) {
  return __builtin_amdgcn_wmma_f32_16x16x32_bf16(false, a, false, b,
                                                 (short)0, c, false, false);
}

__device__ __forceinline__ v16bf make_frag(const __bf16* p0, const __bf16* p1) {
  v8bf lo = *(const v8bf*)p0;
  v8bf hi = *(const v8bf*)p1;
  v16bf r;
#pragma unroll
  for (int i = 0; i < 8; ++i) { r[i] = lo[i]; r[i + 8] = hi[i]; }
  return r;
}

// A fragment: 16(M) x 32(K) bf16. A[i,k] = base[(row0+i)*stride + k].
// ISA layout: lane<16 -> row=lane, K {0..7, 16..23}; lane>=16 -> K {8..15, 24..31}
__device__ __forceinline__ v16bf load_afrag(const __bf16* base, int stride,
                                            int row0, int lane) {
  int r    = row0 + (lane & 15);
  int half = (lane >> 4) * 8;
  const __bf16* p = base + (size_t)r * stride + half;
  return make_frag(p, p + 16);
}

// B fragment: 32(K) x 16(N) bf16. B[k,j] = base[(col0+j)*stride + k].
// ISA layout: lane<16 -> col=lane, K {0..7, 16..23}; lane>=16 -> col=lane-16, K {8..15, 24..31}
__device__ __forceinline__ v16bf load_bfrag(const __bf16* base, int stride,
                                            int col0, int lane) {
  int c    = col0 + (lane & 15);
  int koff = (lane >> 4) * 8;
  const __bf16* p = base + (size_t)c * stride + koff;
  return make_frag(p, p + 16);
}

// ---------------------------------------------------------------------------
// Kernel 1: spectral norm via power iteration (one WG per matrix)
// ---------------------------------------------------------------------------
__global__ void k_spectral(const float* __restrict__ wq,
                           const float* __restrict__ wk,
                           const float* __restrict__ wv,
                           float* __restrict__ inv_sigma) {
  const int id = blockIdx.x;
  const int t  = threadIdx.x;                 // 256 threads
  const float* W;
  int R;
  if (id == 0)      { W = wq; R = 32;  }
  else if (id == 1) { W = wk; R = 32;  }
  else              { W = wv; R = 256; }

  __shared__ float v[256];
  __shared__ float u[256];
  __shared__ float red[256];

  v[t] = 1.0f;
  u[t] = 0.0f;
  __syncthreads();

  float sigma = 1.0f;
  for (int iter = 0; iter < 32; ++iter) {
    // u = W v
    float uu = 0.0f;
    if (t < R) {
      for (int c = 0; c < 256; ++c) uu += W[t * 256 + c] * v[c];
    }
    red[t] = (t < R) ? uu * uu : 0.0f;
    __syncthreads();
    for (int s = 128; s > 0; s >>= 1) {
      if (t < s) red[t] += red[t + s];
      __syncthreads();
    }
    float unorm = sqrtf(red[0]) + 1e-30f;
    __syncthreads();
    if (t < R) u[t] = uu / unorm;
    __syncthreads();
    // v' = W^T u, sigma ~= ||W^T u|| (||u||=1)
    float vv = 0.0f;
    for (int r = 0; r < R; ++r) vv += W[r * 256 + t] * u[r];
    red[t] = vv * vv;
    __syncthreads();
    for (int s = 128; s > 0; s >>= 1) {
      if (t < s) red[t] += red[t + s];
      __syncthreads();
    }
    float vnorm = sqrtf(red[0]) + 1e-30f;
    sigma = vnorm;
    __syncthreads();
    v[t] = vv / vnorm;
    __syncthreads();
  }
  if (t == 0) inv_sigma[id] = 1.0f / sigma;
}

// ---------------------------------------------------------------------------
// Kernel 2: build scaled bf16 weight block Wall[320][256]
//   rows 0..31 = wq/s0, 32..63 = wk/s1, 64..319 = wv/s2
// ---------------------------------------------------------------------------
__global__ void k_build_wall(const float* __restrict__ wq,
                             const float* __restrict__ wk,
                             const float* __restrict__ wv,
                             const float* __restrict__ inv_sigma,
                             __bf16* __restrict__ Wall) {
  int row = blockIdx.x;        // 320
  int t   = threadIdx.x;       // 256
  const float* src;
  float s;
  if (row < 32)      { src = wq + row * 256;        s = inv_sigma[0]; }
  else if (row < 64) { src = wk + (row - 32) * 256; s = inv_sigma[1]; }
  else               { src = wv + (row - 64) * 256; s = inv_sigma[2]; }
  Wall[row * 256 + t] = (__bf16)(src[t] * s);
}

// ---------------------------------------------------------------------------
// Kernel 3a: q/k projections via WMMA. rows 0..63 of Wall @ x(256xN).
//   fT[b][n][32] (rows 0..31), gT[b][n][32] (rows 32..63)
// grid = NB * (NTOK/128), block = 256 (8 waves; wave w owns n-tile w)
// ---------------------------------------------------------------------------
__global__ void k_project_qk(const float* __restrict__ x,
                             const __bf16* __restrict__ Wall,
                             __bf16* __restrict__ fT,
                             __bf16* __restrict__ gT) {
  const int wg = blockIdx.x;
  const int b  = wg >> 5;
  const int n0 = (wg & 31) * 128;
  const int t  = threadIdx.x;
  const int w  = t >> 5;
  const int lane = t & 31;
  const int li = lane & 15, hi = lane >> 4;

  __shared__ __bf16 xs[128 * 32];    // [n_local][k_local], K contiguous

  v8f acc[4];
#pragma unroll
  for (int rt = 0; rt < 4; ++rt)
#pragma unroll
    for (int e = 0; e < 8; ++e) acc[rt][e] = 0.0f;

  for (int k0 = 0; k0 < 256; k0 += 32) {
    __syncthreads();
    // stage x[b, k0..k0+32, n0..n0+128] transposed into LDS (float4 loads)
#pragma unroll
    for (int i = 0; i < 4; ++i) {
      int idx = t + i * 256;
      int kk  = idx >> 5;            // 0..31
      int j4  = (idx & 31) * 4;      // 0..124
      float4 v = *(const float4*)&x[((size_t)b * CIN + (k0 + kk)) * NTOK + n0 + j4];
      xs[(j4 + 0) * 32 + kk] = (__bf16)v.x;
      xs[(j4 + 1) * 32 + kk] = (__bf16)v.y;
      xs[(j4 + 2) * 32 + kk] = (__bf16)v.z;
      xs[(j4 + 3) * 32 + kk] = (__bf16)v.w;
    }
    __syncthreads();

    v16bf bfrag = load_bfrag(xs, 32, w * 16, lane);   // this wave's n-tile
#pragma unroll
    for (int rt = 0; rt < 4; ++rt) {
      v16bf af = load_afrag(Wall + k0, 256, rt * 16, lane);
      acc[rt] = wmma_bf16(af, bfrag, acc[rt]);
    }
  }

#pragma unroll
  for (int rt = 0; rt < 4; ++rt) {
#pragma unroll
    for (int r = 0; r < 8; ++r) {
      int row = rt * 16 + r + 8 * hi;
      int n   = n0 + w * 16 + li;
      __bf16 v = (__bf16)acc[rt][r];
      if (row < 32)
        fT[((size_t)b * NTOK + n) * CQK + row] = v;
      else
        gT[((size_t)b * NTOK + n) * CQK + (row - 32)] = v;
    }
  }
}

// ---------------------------------------------------------------------------
// Kernel 3b: v projection via WMMA. rows 64..319 of Wall @ x(256xN).
//   h[b][c][N], c = Wall row - 64
// grid = NB * (NTOK/64), block = 256 (wave w owns c-tiles {2w,2w+1} x 4 n-tiles)
// ---------------------------------------------------------------------------
__global__ void k_project_v(const float* __restrict__ x,
                            const __bf16* __restrict__ Wall,
                            __bf16* __restrict__ hB) {
  const int wg = blockIdx.x;
  const int b  = wg >> 6;
  const int n0 = (wg & 63) * 64;
  const int t  = threadIdx.x;
  const int w  = t >> 5;
  const int lane = t & 31;
  const int li = lane & 15, hi = lane >> 4;

  __shared__ __bf16 xs[64 * 32];     // [n_local][k_local], K contiguous

  v8f acc[2][4];
#pragma unroll
  for (int a = 0; a < 2; ++a)
#pragma unroll
    for (int c = 0; c < 4; ++c)
#pragma unroll
      for (int e = 0; e < 8; ++e) acc[a][c][e] = 0.0f;

  const __bf16* Wv = Wall + 64 * 256;   // v-weight rows

  for (int k0 = 0; k0 < 256; k0 += 32) {
    __syncthreads();
#pragma unroll
    for (int i = 0; i < 2; ++i) {
      int idx = t + i * 256;
      int kk  = idx >> 4;            // 0..31
      int j4  = (idx & 15) * 4;      // 0..60
      float4 v = *(const float4*)&x[((size_t)b * CIN + (k0 + kk)) * NTOK + n0 + j4];
      xs[(j4 + 0) * 32 + kk] = (__bf16)v.x;
      xs[(j4 + 1) * 32 + kk] = (__bf16)v.y;
      xs[(j4 + 2) * 32 + kk] = (__bf16)v.z;
      xs[(j4 + 3) * 32 + kk] = (__bf16)v.w;
    }
    __syncthreads();

    v16bf bfrag[4];
#pragma unroll
    for (int nt = 0; nt < 4; ++nt) bfrag[nt] = load_bfrag(xs, 32, nt * 16, lane);

#pragma unroll
    for (int cti = 0; cti < 2; ++cti) {
      v16bf af = load_afrag(Wv + k0, 256, (2 * w + cti) * 16, lane);
#pragma unroll
      for (int nt = 0; nt < 4; ++nt)
        acc[cti][nt] = wmma_bf16(af, bfrag[nt], acc[cti][nt]);
    }
  }

#pragma unroll
  for (int cti = 0; cti < 2; ++cti) {
#pragma unroll
    for (int nt = 0; nt < 4; ++nt) {
#pragma unroll
      for (int r = 0; r < 8; ++r) {
        int c = (2 * w + cti) * 16 + r + 8 * hi;
        int n = n0 + nt * 16 + li;
        hB[((size_t)b * CIN + c) * NTOK + n] = (__bf16)acc[cti][nt][r];
      }
    }
  }
}

// ---------------------------------------------------------------------------
// Kernel 4: softmax stats over axis n: M[b,m] = max_n S[n,m], L = sum exp(S-M)
//   S computed on the fly: S(16n x 16m) = A(fT rows n, K=32) x B(gT cols m)
// grid = NB * 32 (each WG covers 128 m; each wave one 16-m tile), block = 256
// ---------------------------------------------------------------------------
__global__ void k_stats(const __bf16* __restrict__ fT,
                        const __bf16* __restrict__ gT,
                        float* __restrict__ Mmax,
                        float* __restrict__ Lsum) {
  const int wg = blockIdx.x;
  const int b    = wg >> 5;
  const int mblk = wg & 31;
  const int t = threadIdx.x;
  const int w = t >> 5, lane = t & 31;
  const int li = lane & 15, hi = lane >> 4;
  const int m0 = mblk * 128 + w * 16;

  const __bf16* fTb = fT + (size_t)b * NTOK * CQK;
  const __bf16* gTb = gT + (size_t)b * NTOK * CQK;

  v16bf gfrag = load_bfrag(gTb, CQK, m0, lane);   // fixed query columns

  float mx = -1e30f, s = 0.0f;
  for (int n0 = 0; n0 < NTOK; n0 += 16) {
    v16bf af = load_afrag(fTb, CQK, n0, lane);
    v8f z = {0.f, 0.f, 0.f, 0.f, 0.f, 0.f, 0.f, 0.f};
    v8f sv = wmma_bf16(af, gfrag, z);
    // lane owns column m0+li, rows (r + 8*hi): online logsumexp (lane-local)
#pragma unroll
    for (int r = 0; r < 8; ++r) {
      float v = sv[r];
      if (v > mx) { s = s * __expf(mx - v) + 1.0f; mx = v; }
      else        { s += __expf(v - mx); }
    }
  }
  // combine the two half-columns (lanes L and L^16 share column m0+li)
  float omx = __shfl_xor(mx, 16, 32);
  float os  = __shfl_xor(s, 16, 32);
  float nm  = fmaxf(mx, omx);
  float ns  = s * __expf(mx - nm) + os * __expf(omx - nm);
  if (hi == 0) {
    Mmax[(size_t)b * NTOK + m0 + li] = nm;
    Lsum[(size_t)b * NTOK + m0 + li] = ns;
  }
}

// ---------------------------------------------------------------------------
// Kernel 5: fused output pass, double-buffered P tile (one barrier per step).
//   For m-block of 64: loop n in steps of 32:
//     waves 0..7 each compute one 16x16 S tile (2 n-tiles x 4 m-tiles),
//     P = exp(S - M[m]) -> LDS [64 m][32 n] bf16 (ping-pong),
//     every wave accumulates its 2 c-tiles x 4 m-tiles: acc += h @ P.
//   Epilogue: out = gamma * acc / L[m] + x.
// grid = NB * 64, block = 256
// ---------------------------------------------------------------------------
__global__ void k_attn_out(const float* __restrict__ x,
                           const __bf16* __restrict__ fT,
                           const __bf16* __restrict__ gT,
                           const __bf16* __restrict__ hB,
                           const float* __restrict__ Mmax,
                           const float* __restrict__ Lsum,
                           const float* __restrict__ gamma_p,
                           float* __restrict__ out) {
  const int wg = blockIdx.x;
  const int b  = wg >> 6;
  const int m0 = (wg & 63) * 64;
  const int t  = threadIdx.x;
  const int w  = t >> 5, lane = t & 31;
  const int li = lane & 15, hi = lane >> 4;

  const __bf16* fTb = fT + (size_t)b * NTOK * CQK;
  const __bf16* gTb = gT + (size_t)b * NTOK * CQK;
  const __bf16* hb  = hB + (size_t)b * CIN * NTOK;

  __shared__ __bf16 Plds[2][64 * 32];   // ping-pong [m_local][n_local]

  // score role: wave w computes S tile (n-tile s_nt, m-tile s_mt)
  const int s_nt = w >> 2;           // 0..1
  const int s_mt = w & 3;            // 0..3
  const float s_mmax = Mmax[(size_t)b * NTOK + m0 + s_mt * 16 + li];
  v16bf gfrag = load_bfrag(gTb, CQK, m0 + s_mt * 16, lane);
  const int p_wr = (s_mt * 16 + li) * 32 + s_nt * 16 + 8 * hi;

  // output role: wave w owns c-tiles {2w, 2w+1} x 4 m-tiles
  const int ct0 = w * 2;
  float invL[4];
#pragma unroll
  for (int mt = 0; mt < 4; ++mt)
    invL[mt] = 1.0f / Lsum[(size_t)b * NTOK + m0 + mt * 16 + li];

  v8f acc[2][4];
#pragma unroll
  for (int c = 0; c < 2; ++c)
#pragma unroll
    for (int mt = 0; mt < 4; ++mt)
#pragma unroll
      for (int e = 0; e < 8; ++e) acc[c][mt][e] = 0.0f;

  // prologue: compute P for n0 = 0 into buffer 0
  {
    v16bf af = load_afrag(fTb, CQK, s_nt * 16, lane);
    v8f z = {0.f, 0.f, 0.f, 0.f, 0.f, 0.f, 0.f, 0.f};
    v8f sv = wmma_bf16(af, gfrag, z);
#pragma unroll
    for (int r = 0; r < 8; ++r)
      Plds[0][p_wr + r] = (__bf16)__expf(sv[r] - s_mmax);
  }
  __syncthreads();

  int cur = 0;
  for (int n0 = 0; n0 < NTOK; n0 += 32) {
    // consume current P buffer
    v16bf pb[4];
#pragma unroll
    for (int mt = 0; mt < 4; ++mt)
      pb[mt] = load_bfrag(&Plds[cur][0], 32, mt * 16, lane);

    // produce next P buffer (overlaps with accumulate WMMAs below)
    if (n0 + 32 < NTOK) {
      v16bf af = load_afrag(fTb, CQK, n0 + 32 + s_nt * 16, lane);
      v8f z = {0.f, 0.f, 0.f, 0.f, 0.f, 0.f, 0.f, 0.f};
      v8f sv = wmma_bf16(af, gfrag, z);
#pragma unroll
      for (int r = 0; r < 8; ++r)
        Plds[cur ^ 1][p_wr + r] = (__bf16)__expf(sv[r] - s_mmax);
    }

    // acc += h(16c x 32n) @ P(32n x 16m)
#pragma unroll
    for (int cti = 0; cti < 2; ++cti) {
      v16bf ha = load_afrag(hb + n0, NTOK, (ct0 + cti) * 16, lane);
#pragma unroll
      for (int mt = 0; mt < 4; ++mt)
        acc[cti][mt] = wmma_bf16(ha, pb[mt], acc[cti][mt]);
    }

    __syncthreads();
    cur ^= 1;
  }

  const float gma = gamma_p[0];
#pragma unroll
  for (int cti = 0; cti < 2; ++cti) {
#pragma unroll
    for (int mt = 0; mt < 4; ++mt) {
#pragma unroll
      for (int r = 0; r < 8; ++r) {
        int c = (ct0 + cti) * 16 + r + 8 * hi;
        int m = m0 + mt * 16 + li;
        size_t idx = ((size_t)b * CIN + c) * NTOK + m;
        out[idx] = gma * acc[cti][mt][r] * invL[mt] + x[idx];
      }
    }
  }
}

// ---------------------------------------------------------------------------
// Host launch
// ---------------------------------------------------------------------------
extern "C" void kernel_launch(void* const* d_in, const int* in_sizes, int n_in,
                              void* d_out, int out_size, void* d_ws, size_t ws_size,
                              hipStream_t stream) {
  (void)in_sizes; (void)n_in; (void)out_size; (void)ws_size;
  const float* x  = (const float*)d_in[0];
  const float* wq = (const float*)d_in[1];
  const float* wk = (const float*)d_in[2];
  const float* wv = (const float*)d_in[3];
  const float* gm = (const float*)d_in[4];
  float* out = (float*)d_out;

  // workspace layout (~10.8 MB)
  float*  inv_sigma = (float*)d_ws;                               // 3 f32 (pad 256B)
  __bf16* Wall = (__bf16*)((char*)d_ws + 256);                    // 320*256
  __bf16* fT   = Wall + 320 * 256;                                // NB*NTOK*32
  __bf16* gT   = fT + (size_t)NB * NTOK * CQK;
  __bf16* hB   = gT + (size_t)NB * NTOK * CQK;                    // NB*CIN*NTOK
  float*  Mm   = (float*)(hB + (size_t)NB * CIN * NTOK);          // NB*NTOK
  float*  Ls   = Mm + (size_t)NB * NTOK;

  k_spectral  <<<3,       256, 0, stream>>>(wq, wk, wv, inv_sigma);
  k_build_wall<<<320,     256, 0, stream>>>(wq, wk, wv, inv_sigma, Wall);
  k_project_qk<<<NB * 32, 256, 0, stream>>>(x, Wall, fT, gT);
  k_project_v <<<NB * 64, 256, 0, stream>>>(x, Wall, hB);
  k_stats     <<<NB * 32, 256, 0, stream>>>(fT, gT, Mm, Ls);
  k_attn_out  <<<NB * 64, 256, 0, stream>>>(x, fT, gT, hB, Mm, Ls, gm, out);
}
